// VectorAttention_20572893348692
// MI455X (gfx1250) — matrix operand order repaired and checked by reading, not compiled, not run on hardware
//
#include <hip/hip_runtime.h>
#include <hip/hip_bf16.h>
#include <math.h>

typedef __attribute__((ext_vector_type(16))) _Float16 v16h;
typedef __attribute__((ext_vector_type(8)))  _Float16 v8h;
typedef __attribute__((ext_vector_type(8)))  float    v8f;

#define Nn 512
#define Dd 32
#define Hh 64
#define Bb 2
#define NNP (Nn*Nn)          // pairs per batch
#define TILES (NNP/16)       // 16-pair tiles per batch
#define NW 4                 // waves per block (128 threads, wave32)

union AV { v16h v; v8h h[2]; };

// ---------------------------------------------------------------------------
// Pass A (SCORE=true): compute scores for every pair, write to ws.
// Pass C (SCORE=false): recompute joined, apply softmax weights, reduce to out.
// One wave handles one 16-pair tile. All 16 pairs of a tile share the same i
// (row index), j = jbase..jbase+15 (tiles are 16-aligned, N % 16 == 0).
// ---------------------------------------------------------------------------
template<bool SCORE>
__global__ __launch_bounds__(128)
void pair_kernel(const float* __restrict__ pos,  const float* __restrict__ vals,
                 const float* __restrict__ W1,   const float* __restrict__ b1,
                 const float* __restrict__ W2,   const float* __restrict__ b2,
                 const float* __restrict__ Ws1,  const float* __restrict__ bs1,
                 const float* __restrict__ Ws2,  const float* __restrict__ bs2,
                 float* __restrict__ scores,     const float* __restrict__ stats,
                 float* __restrict__ out)
{
    // Weights staged in LDS, pre-swizzled into the WMMA B-matrix lane layout:
    //   B 32x16 f16 tile: lane 0-15 -> K={0..7,16..23} (2 halfs/VGPR), lane 16-31 -> K={8..15,24..31},
    //   column = lane & 15.  Stored as [tile][lane][16 halfs] so each operand is a
    //   contiguous 32B per lane.
    __shared__ alignas(32) _Float16 w2b [2][2][32][16];   // [ktile][ntile][lane][j]
    __shared__ alignas(32) _Float16 ws1b[4][32][16];      // [ntile][lane][j] (K=32)
    __shared__ alignas(32) _Float16 h1buf[NW][16][Hh];    // per-wave hidden1 (A matrix src)
    __shared__ alignas(32) _Float16 jbuf [NW][16][Dd];    // per-wave joined  (A matrix src)
    __shared__ float w1s[2*Hh];
    __shared__ float b1s[Hh], bs1s[Hh], ws2s[Hh], b2s[Dd];
    __shared__ float bacc[Dd];

    const int tid = threadIdx.x;

    for (int idx = tid; idx < 2*2*32*16; idx += blockDim.x) {
        int kt = idx >> 10, nt = (idx >> 9) & 1, ln = (idx >> 4) & 31, j = idx & 15;
        int K = kt*32 + ((ln < 16) ? 0 : 8) + ((j < 8) ? j : j + 8);
        int n = nt*16 + (ln & 15);
        (&w2b[0][0][0][0])[idx] = (_Float16)W2[K*Dd + n];          // W2: (64,32)
    }
    for (int idx = tid; idx < 4*32*16; idx += blockDim.x) {
        int nt = idx >> 9, ln = (idx >> 4) & 31, j = idx & 15;
        int K = ((ln < 16) ? 0 : 8) + ((j < 8) ? j : j + 8);
        int n = nt*16 + (ln & 15);
        (&ws1b[0][0][0])[idx] = (_Float16)Ws1[K*Hh + n];           // Ws1: (32,64)
    }
    for (int idx = tid; idx < Hh; idx += blockDim.x) {
        w1s[idx]      = W1[idx];
        w1s[Hh + idx] = W1[Hh + idx];
        b1s[idx]  = b1[idx];
        bs1s[idx] = bs1[idx];
        ws2s[idx] = Ws2[idx];
        if (idx < Dd) b2s[idx] = b2[idx];
    }
    if (!SCORE && tid < Dd) bacc[tid] = 0.f;
    __syncthreads();

    const int w    = tid >> 5;
    const int lane = tid & 31;
    const int tileg = blockIdx.x * NW + w;
    const int batch = tileg / TILES;
    const int tb    = tileg % TILES;
    const int pbase = tb * 16;
    const int i_idx = pbase >> 9;          // shared row index for the tile
    const int jbase = pbase & (Nn - 1);
    const int r     = lane & 15;           // pair-row this lane feeds / reads
    const int hs    = lane >> 4;           // which half of the lane group
    const int coln  = lane & 15;           // C/D column within an n-tile

    const float* posB = pos  + (size_t)batch * Nn * 3;
    const float* valB = vals + (size_t)batch * Nn * Dd;

    // ---- invariants for pair (i_idx, jbase+r): a = pos[j], b = pos[i] ----
    float ax = posB[(jbase + r)*3 + 0], ay = posB[(jbase + r)*3 + 1], az = posB[(jbase + r)*3 + 2];
    float bx = posB[i_idx*3 + 0],       by = posB[i_idx*3 + 1],       bz = posB[i_idx*3 + 2];
    float inv0 = ax*bx + ay*by + az*bz;
    float e12 = ax*by - ay*bx, e13 = ax*bz - az*bx, e23 = ay*bz - az*by;
    float inv1 = sqrtf(e12*e12 + e13*e13 + e23*e23);

    // ---- layer 1: h1 = relu(invar @ W1 + b1), each lane does 32 of 64 ----
    const int hb = hs * 32;
    #pragma unroll
    for (int h = 0; h < 32; ++h) {
        int hh = hb + h;
        float v = fmaxf(inv0 * w1s[hh] + inv1 * w1s[Hh + hh] + b1s[hh], 0.f);
        h1buf[w][r][hh] = (_Float16)v;
    }

    // ---- GEMM1: (16x64 h1) @ (64x32 W2): 2 K-tiles x 2 N-tiles of WMMA ----
    // A layout per ISA: lane 0-15 row=lane holds K = {0..7, 16..23}+kt*32,
    //                   lane 16-31 same row holds K = {8..15, 24..31}+kt*32.
    const int kbA = hs * 8;
    v8f c0 = {}, c1 = {};
    #pragma unroll
    for (int kt = 0; kt < 2; ++kt) {
        AV a;
        a.h[0] = *(const v8h*)&h1buf[w][r][kt*32 + kbA];
        a.h[1] = *(const v8h*)&h1buf[w][r][kt*32 + kbA + 16];
        v16h bm0 = *(const v16h*)&w2b[kt][0][lane][0];
        v16h bm1 = *(const v16h*)&w2b[kt][1][lane][0];
        c0 = __builtin_amdgcn_wmma_f32_16x16x32_f16(false, a.v, false, bm0, (short)0, c0, false, false);
        c1 = __builtin_amdgcn_wmma_f32_16x16x32_f16(false, a.v, false, bm1, (short)0, c1, false, false);
    }

    // ---- joined = 0.5*(invar_values + b2) + 0.25*(vals[i] + vals[j]) ----
    // D layout: element k of v8f is row k + 8*hs, column nt*16 + coln.
    float jf[2][8];
    #pragma unroll
    for (int nt = 0; nt < 2; ++nt) {
        int col = nt*16 + coln;
        float vi = valB[i_idx*Dd + col];
        v8f cc = nt ? c1 : c0;
        #pragma unroll
        for (int k = 0; k < 8; ++k) {
            int row = k + hs*8;
            float jv = 0.5f*(cc[k] + b2s[col]) + 0.25f*(vi + valB[(jbase + row)*Dd + col]);
            jf[nt][k] = jv;
            if (SCORE) jbuf[w][row][col] = (_Float16)jv;
        }
    }

    if (SCORE) {
        // ---- GEMM2: (16x32 joined) @ (32x64 Ws1): K=32, 4 N-tiles ----
        AV a2;
        a2.h[0] = *(const v8h*)&jbuf[w][r][kbA];
        a2.h[1] = *(const v8h*)&jbuf[w][r][kbA + 16];
        float part[8] = {0.f,0.f,0.f,0.f,0.f,0.f,0.f,0.f};
        #pragma unroll
        for (int nt = 0; nt < 4; ++nt) {
            v16h bm = *(const v16h*)&ws1b[nt][lane][0];
            v8f d = {};
            d = __builtin_amdgcn_wmma_f32_16x16x32_f16(false, a2.v, false, bm, (short)0, d, false, false);
            int col = nt*16 + coln;
            float wv = ws2s[col], bb = bs1s[col];
            #pragma unroll
            for (int k = 0; k < 8; ++k)
                part[k] += fmaxf(d[k] + bb, 0.f) * wv;      // relu(h2)·Ws2
        }
        // reduce over the 16 columns held by this lane-half
        #pragma unroll
        for (int m = 8; m >= 1; m >>= 1)
            #pragma unroll
            for (int k = 0; k < 8; ++k)
                part[k] += __shfl_xor(part[k], m, 32);
        if (coln == 0) {
            float bsv = bs2[0];
            #pragma unroll
            for (int k = 0; k < 8; ++k)
                scores[(size_t)batch*NNP + pbase + k + hs*8] = part[k] + bsv;
        }
    } else {
        // ---- weighted accumulation with precomputed softmax stats ----
        float mx = stats[batch*2 + 0], isum = stats[batch*2 + 1];
        float acc0 = 0.f, acc1 = 0.f;
        #pragma unroll
        for (int k = 0; k < 8; ++k) {
            int row = k + hs*8;
            float s  = scores[(size_t)batch*NNP + pbase + row];
            float wv = __expf(s - mx) * isum;
            acc0 += wv * jf[0][k];
            acc1 += wv * jf[1][k];
        }
        __syncthreads();
        atomicAdd(&bacc[coln],      acc0);
        atomicAdd(&bacc[16 + coln], acc1);
        __syncthreads();
        if (tid < Dd) atomicAdd(&out[batch*Dd + tid], bacc[tid]);
    }
}

// ---------------------------------------------------------------------------
// Per-batch softmax statistics: max and 1/sum(exp(s-max)). One block / batch.
// ---------------------------------------------------------------------------
__global__ __launch_bounds__(256)
void softmax_stats_kernel(const float* __restrict__ scores, float* __restrict__ stats)
{
    const int batch = blockIdx.x;
    const float* s = scores + (size_t)batch * NNP;
    __shared__ float red[256];
    const int tid = threadIdx.x;

    float mx = -1e30f;
    for (int idx = tid; idx < NNP; idx += 256) mx = fmaxf(mx, s[idx]);
    red[tid] = mx; __syncthreads();
    for (int off = 128; off > 0; off >>= 1) {
        if (tid < off) red[tid] = fmaxf(red[tid], red[tid + off]);
        __syncthreads();
    }
    mx = red[0]; __syncthreads();

    float sum = 0.f;
    for (int idx = tid; idx < NNP; idx += 256) sum += __expf(s[idx] - mx);
    red[tid] = sum; __syncthreads();
    for (int off = 128; off > 0; off >>= 1) {
        if (tid < off) red[tid] += red[tid + off];
        __syncthreads();
    }
    if (tid == 0) { stats[batch*2 + 0] = mx; stats[batch*2 + 1] = 1.0f / red[0]; }
}

__global__ void zero_out_kernel(float* out) { out[threadIdx.x] = 0.f; }

// ---------------------------------------------------------------------------
extern "C" void kernel_launch(void* const* d_in, const int* in_sizes, int n_in,
                              void* d_out, int out_size, void* d_ws, size_t ws_size,
                              hipStream_t stream)
{
    const float* pos = (const float*)d_in[0];
    const float* val = (const float*)d_in[1];
    const float* W1  = (const float*)d_in[2];
    const float* b1  = (const float*)d_in[3];
    const float* W2  = (const float*)d_in[4];
    const float* b2  = (const float*)d_in[5];
    const float* Ws1 = (const float*)d_in[6];
    const float* bs1 = (const float*)d_in[7];
    const float* Ws2 = (const float*)d_in[8];
    const float* bs2 = (const float*)d_in[9];

    float* scores = (float*)d_ws;                       // B*N*N floats = 2 MB
    float* stats  = scores + (size_t)Bb * NNP;          // 4 floats
    float* out    = (float*)d_out;                      // (B, D) = 64 floats

    const int blocks = (Bb * TILES) / NW;               // 8192 blocks of 128

    zero_out_kernel<<<1, Bb*Dd, 0, stream>>>(out);
    pair_kernel<true ><<<blocks, NW*32, 0, stream>>>(pos, val, W1, b1, W2, b2,
                                                     Ws1, bs1, Ws2, bs2,
                                                     scores, stats, out);
    softmax_stats_kernel<<<Bb, 256, 0, stream>>>(scores, stats);
    pair_kernel<false><<<blocks, NW*32, 0, stream>>>(pos, val, W1, b1, W2, b2,
                                                     Ws1, bs1, Ws2, bs2,
                                                     scores, stats, out);
}
